// ExpertMLP_12043088298373
// MI455X (gfx1250) — compile-verified, matrix operand order checked by reading, not run on hardware
//
#include <hip/hip_runtime.h>
#include <hip/hip_bf16.h>
#include <math.h>

#define E_   32
#define CAP_ 4096
#define D_   256
#define HD_  1024

typedef __attribute__((ext_vector_type(16))) __bf16 v16bf;
typedef __attribute__((ext_vector_type(8)))  float  v8f;

union Frag {
  uint4 u[2];
  v16bf v;
};

// round-to-nearest-even fp32 -> bf16 (bit trick)
__device__ __forceinline__ unsigned short f2bf(float a) {
  unsigned int ua = __float_as_uint(a);
  return (unsigned short)((ua + 0x7FFFu + ((ua >> 16) & 1u)) >> 16);
}
__device__ __forceinline__ unsigned int pack2bf(float a, float b) {
  return (unsigned int)f2bf(a) | ((unsigned int)f2bf(b) << 16);
}
__device__ __forceinline__ float gelu_exact(float x) {
  return 0.5f * x * (1.0f + erff(x * 0.70710678118654752440f));
}

// ---------------------------------------------------------------------------
// Prep: convert fp32 weights to bf16, packed per 32(K)x16(N) WMMA-B tile.
// Tile = 1024 bytes: lane L stores 32 bytes at L*32:
//   bytes  0-15 : K = k0 + sel + {0..7}   of column n0 + (L%16)
//   bytes 16-31 : K = k0 + sel + 16 + {0..7}
//   sel = 0 for L<16, 8 for L>=16   (matches ISA 16-bit B layout)
// W1 packed: [E][64 ntiles][8 ktiles]   (16 MB)
// W2 packed: [E][16 ntiles][32 ktiles]  (16 MB)
// ---------------------------------------------------------------------------
__global__ void __launch_bounds__(256) prep_weights_kernel(
    const float* __restrict__ w_fc, const float* __restrict__ w_proj,
    unsigned int* __restrict__ w1p, unsigned int* __restrict__ w2p) {
  const int wave = threadIdx.x >> 5;
  const int lane = threadIdx.x & 31;
  const int t = blockIdx.x * 8 + wave;          // 32768 tiles total
  const int sel = (lane >> 4) * 8;

  if (t < E_ * 512) {                           // W1: [256 x 1024], ldn = 1024
    const int e = t >> 9, rem = t & 511;
    const int nt = rem >> 3, kt = rem & 7;
    const float* src = w_fc + (size_t)e * D_ * HD_;
    const int col = nt * 16 + (lane & 15);
    unsigned int* dst = w1p + (size_t)t * 256 + lane * 8;
#pragma unroll
    for (int d = 0; d < 8; ++d) {
      const int part = d >> 2, dd = d & 3;
      const int k = kt * 32 + part * 16 + sel + dd * 2;
      dst[d] = pack2bf(src[(size_t)k * HD_ + col], src[(size_t)(k + 1) * HD_ + col]);
    }
  } else {                                      // W2: [1024 x 256], ldn = 256
    const int t2 = t - E_ * 512;
    const int e = t2 >> 9, rem = t2 & 511;
    const int nt = rem >> 5, kt = rem & 31;
    const float* src = w_proj + (size_t)e * HD_ * D_;
    const int col = nt * 16 + (lane & 15);
    unsigned int* dst = w2p + (size_t)t2 * 256 + lane * 8;
#pragma unroll
    for (int d = 0; d < 8; ++d) {
      const int part = d >> 2, dd = d & 3;
      const int k = kt * 32 + part * 16 + sel + dd * 2;
      dst[d] = pack2bf(src[(size_t)k * D_ + col], src[(size_t)(k + 1) * D_ + col]);
    }
  }
}

// ---------------------------------------------------------------------------
// Fused expert MLP: out = gelu(x @ w_fc) @ w_proj, 64 rows per workgroup.
// Hidden dim processed in 8 chunks of 128 columns; 2x2 / 2x4 register
// blocking per wave so every WMMA costs ~1 global b128 + ~0.75 ds b128.
// ---------------------------------------------------------------------------
__global__ void __launch_bounds__(256, 1) expert_mlp_kernel(
    const float* __restrict__ x,
    const unsigned int* __restrict__ w1p,
    const unsigned int* __restrict__ w2p,
    float* __restrict__ out) {
  __shared__ unsigned short Xs[64 * 264];  // 64 rows x 256 K, bf16, padded stride
  __shared__ unsigned short Hs[64 * 136];  // 64 rows x 128 hidden cols, bf16, padded

  const int tid  = threadIdx.x;
  const int wave = tid >> 5;
  const int lane = tid & 31;
  const int e  = blockIdx.x >> 6;          // 64 row-blocks per expert
  const int m0 = (blockIdx.x & 63) << 6;

  // ---- load + convert X tile [64 x 256] fp32 -> bf16 LDS (coalesced) ----
  const float4* xt = (const float4*)(x + ((size_t)e * CAP_ + m0) * D_);
#pragma unroll
  for (int it = 0; it < 16; ++it) {
    const int f = it * 256 + tid;
    const int row = f >> 6, c4 = f & 63;
    const float4 v = xt[row * 64 + c4];
    uint2 p;
    p.x = pack2bf(v.x, v.y);
    p.y = pack2bf(v.z, v.w);
    *(uint2*)&Xs[row * 264 + c4 * 4] = p;
  }
  __syncthreads();

  // wave work mapping (2x2 blocking phase 1, 2x4 blocking phase 2)
  const int mtg = wave & 1;                // phase-1 row tiles {2*mtg, 2*mtg+1}
  const int ctg = wave >> 1;               // phase-1 col tiles {2*ctg, 2*ctg+1} (chunk-local)
  const int mtg2 = wave & 1;               // phase-2 row tiles {2*mtg2, 2*mtg2+1}
  const int ntg2 = wave >> 1;              // phase-2 col tiles {4*ntg2 .. +3}

  const unsigned int* w1e = w1p + (size_t)e * 512 * 256;
  const unsigned int* w2e = w2p + (size_t)e * 512 * 256;

  const int laneRow = lane & 15;
  const int laneSel = (lane >> 4) << 3;    // 0 or 8 halves (A/B fragment K select)
  const int rOff    = (lane >> 4) << 3;    // C-layout row offset (0 or 8)

  v8f acc2[2][4] = {};                     // persistent Out accumulators

  for (int c = 0; c < 8; ++c) {
    // ---------------- phase 1: Hchunk[64x128] = Xtile @ W1[:, c*128 .. +128]
    v8f acc1[2][2] = {};
#pragma unroll
    for (int kt = 0; kt < 8; ++kt) {
      Frag a[2], b[2];
#pragma unroll
      for (int i = 0; i < 2; ++i) {
        const unsigned short* ap =
            &Xs[((mtg * 2 + i) * 16 + laneRow) * 264 + kt * 32 + laneSel];
        a[i].u[0] = *(const uint4*)ap;
        a[i].u[1] = *(const uint4*)(ap + 16);
      }
#pragma unroll
      for (int j = 0; j < 2; ++j) {
        const int nt = c * 8 + ctg * 2 + j;
        const uint4* bp = ((const uint4*)(w1e + (size_t)(nt * 8 + kt) * 256)) + lane * 2;
        b[j].u[0] = bp[0];
        b[j].u[1] = bp[1];
      }
#pragma unroll
      for (int i = 0; i < 2; ++i)
#pragma unroll
        for (int j = 0; j < 2; ++j)
          acc1[i][j] = __builtin_amdgcn_wmma_f32_16x16x32_bf16(
              false, a[i].v, false, b[j].v, (short)0, acc1[i][j], false, false);
    }
    // GELU (exact erf), convert, write H chunk to LDS (C-layout -> row-major)
#pragma unroll
    for (int i = 0; i < 2; ++i) {
      const int rbase = (mtg * 2 + i) * 16 + rOff;
#pragma unroll
      for (int j = 0; j < 2; ++j) {
        const int col = (ctg * 2 + j) * 16 + laneRow;
#pragma unroll
        for (int v = 0; v < 8; ++v)
          Hs[(rbase + v) * 136 + col] = f2bf(gelu_exact(acc1[i][j][v]));
      }
    }
    __syncthreads();

    // ---------------- phase 2: Out += Hchunk @ W2[c*128 .. +128, :]
#pragma unroll
    for (int kt2 = 0; kt2 < 4; ++kt2) {
      const int ktg = c * 4 + kt2;
      Frag a[2], b[4];
#pragma unroll
      for (int i = 0; i < 2; ++i) {
        const unsigned short* ap =
            &Hs[((mtg2 * 2 + i) * 16 + laneRow) * 136 + kt2 * 32 + laneSel];
        a[i].u[0] = *(const uint4*)ap;
        a[i].u[1] = *(const uint4*)(ap + 16);
      }
#pragma unroll
      for (int n = 0; n < 4; ++n) {
        const int nt = ntg2 * 4 + n;
        const uint4* bp = ((const uint4*)(w2e + (size_t)(nt * 32 + ktg) * 256)) + lane * 2;
        b[n].u[0] = bp[0];
        b[n].u[1] = bp[1];
      }
#pragma unroll
      for (int i = 0; i < 2; ++i)
#pragma unroll
        for (int n = 0; n < 4; ++n)
          acc2[i][n] = __builtin_amdgcn_wmma_f32_16x16x32_bf16(
              false, a[i].v, false, b[n].v, (short)0, acc2[i][n], false, false);
    }
    __syncthreads();
  }

  // ---- write Out tile [64 x 256] fp32 ----
  float* ot = out + ((size_t)e * CAP_ + m0) * D_;
#pragma unroll
  for (int i = 0; i < 2; ++i) {
    const int rbase = (mtg2 * 2 + i) * 16 + rOff;
#pragma unroll
    for (int n = 0; n < 4; ++n) {
      const int col = (ntg2 * 4 + n) * 16 + laneRow;
#pragma unroll
      for (int v = 0; v < 8; ++v)
        ot[(size_t)(rbase + v) * D_ + col] = acc2[i][n][v];
    }
  }
}

extern "C" void kernel_launch(void* const* d_in, const int* in_sizes, int n_in,
                              void* d_out, int out_size, void* d_ws, size_t ws_size,
                              hipStream_t stream) {
  const float* x      = (const float*)d_in[0];
  const float* w_fc   = (const float*)d_in[1];
  const float* w_proj = (const float*)d_in[2];
  float* out          = (float*)d_out;

  // workspace: 16 MB packed-bf16 W1 + 16 MB packed-bf16 W2
  unsigned int* w1p = (unsigned int*)d_ws;
  unsigned int* w2p = w1p + (size_t)E_ * 512 * 256;

  // 32768 tiles, 8 tiles (waves) per 256-thread block
  prep_weights_kernel<<<4096, 256, 0, stream>>>(w_fc, w_proj, w1p, w2p);

  // 32 experts x 64 row-blocks
  expert_mlp_kernel<<<E_ * (CAP_ / 64), 256, 0, stream>>>(x, w1p, w2p, out);
}
